// ConvEGNN_44822278701376
// MI455X (gfx1250) — matrix operand order repaired and checked by reading, not compiled, not run on hardware
//
#include <hip/hip_runtime.h>
#include <math.h>

#define BSZ   32
#define NNODE 512
#define KNB   32
#define HH    64
#define WAVES 8
#define NODES (BSZ * NNODE)

// ---- LDS layout (dynamic shared memory, byte offsets) ----
#define OFF_WE1F  0        // 4kc*4nt*32*16 bf16 = 16384 B (fragment order)
#define OFF_WE2F  16384    // 2kc*4nt*32*16 bf16 =  8192 B
#define OFF_WX1F  24576    //                       8192 B
#define OFF_WH1T  32768    // [64 n][128 k] bf16 = 16384 B (transposed)
#define OFF_WH2T  49152    // [64 n][64 k]  bf16 =  8192 B
#define OFF_F32   57344    // 644 f32 = 2576 B
#define OFF_WAVE  59936    // per-wave regions (32B aligned)
#define PERWAVE   5504     // buf 4096 | hibf 128 | hi 256 | dist 128 | mi 256 | t1 256 | red 384
#define SMEM_TOTAL (OFF_WAVE + WAVES * PERWAVE)   // 103968 B (< 320 KB/WGP)

typedef __attribute__((ext_vector_type(16))) __bf16          v16bf;
typedef __attribute__((ext_vector_type(8)))  float           v8f;
typedef __attribute__((ext_vector_type(8)))  unsigned short  v8us;
typedef __attribute__((ext_vector_type(4)))  unsigned short  v4us;

union BF16x16 { v16bf v; unsigned short s[16]; };

__device__ __forceinline__ unsigned short f2bf(float f) {
  union { float f; unsigned u; } x; x.f = f;
  return (unsigned short)((x.u + 0x7FFFu + ((x.u >> 16) & 1u)) >> 16);
}
__device__ __forceinline__ float bf2f(unsigned short h) {
  union { unsigned u; float f; } x; x.u = ((unsigned)h) << 16;
  return x.f;
}
__device__ __forceinline__ float silu(float x) { return x / (1.f + expf(-x)); }

// A-fragment layout address (ushort units) for a 32x64 bf16 edge tile:
// element (edge M, feature f) lives at ((mt*2+fc)*32 + slot)*16 + i,
// slot = hp*16 + (M&15);  A-layout: K = (i/8)*16 + i%8 + 8*hp  (ISA 16-bit A 16x32)
__device__ __forceinline__ int fragAddr(int edge, int f) {
  const int mt = edge >> 4, fc = f >> 5, kk = f & 31;
  const int hp = (kk >> 3) & 1;
  const int ii = ((kk >> 4) << 3) + (kk & 7);
  const int slot = hp * 16 + (edge & 15);
  return ((mt * 2 + fc) * 32 + slot) * 16 + ii;
}

__global__ __launch_bounds__(256) void egnn_fused(
    const float* __restrict__ ped, const float* __restrict__ hst,
    const float* __restrict__ hneigh, const float* __restrict__ rela,
    const unsigned char* __restrict__ mask,
    const float* __restrict__ we1w, const float* __restrict__ we1b,
    const float* __restrict__ we2w, const float* __restrict__ we2b,
    const float* __restrict__ wx1w, const float* __restrict__ wx1b,
    const float* __restrict__ wx2w, const float* __restrict__ wx2b,
    const float* __restrict__ waw,  const float* __restrict__ wab,
    const float* __restrict__ wh1w, const float* __restrict__ wh1b,
    const float* __restrict__ wh2w, const float* __restrict__ wh2b,
    float* __restrict__ out6, float* __restrict__ hout)
{
  extern __shared__ char smem[];
  unsigned short* we1f  = (unsigned short*)(smem + OFF_WE1F);
  unsigned short* we2f  = (unsigned short*)(smem + OFF_WE2F);
  unsigned short* wx1f  = (unsigned short*)(smem + OFF_WX1F);
  unsigned short* wh1Ts = (unsigned short*)(smem + OFF_WH1T);
  unsigned short* wh2Ts = (unsigned short*)(smem + OFF_WH2T);
  float* fs    = (float*)(smem + OFF_F32);
  float* we1bs = fs + 0;   float* we2bs = fs + 64;  float* wx1bs = fs + 128;
  float* wh1bs = fs + 192; float* wh2bs = fs + 256;
  float* wx2s  = fs + 320; float* was   = fs + 448;
  float* we1ls = fs + 576;                      // we1 row 128 (dist column), f32
  float* wx2bs = fs + 640; float* wabs  = fs + 642;

  const int tid = threadIdx.x;

  // ---- stage weights fp32 -> bf16 LDS in WMMA B-fragment order ----
  for (int o = tid; o < 8192; o += 256) {       // we1 rows 0..127
    const int i = o & 15, s = (o >> 4) & 31, nt = (o >> 9) & 3, kc = (o >> 11) & 3;
    const int K = kc * 32 + ((i >> 3) << 4) + ((s >> 4) << 3) + (i & 7);
    const int N = nt * 16 + (s & 15);
    we1f[o] = f2bf(we1w[K * 64 + N]);
  }
  for (int o = tid; o < 4096; o += 256) {       // we2, wx1
    const int i = o & 15, s = (o >> 4) & 31, nt = (o >> 9) & 3, kc = o >> 11;
    const int K = kc * 32 + ((i >> 3) << 4) + ((s >> 4) << 3) + (i & 7);
    const int N = nt * 16 + (s & 15);
    we2f[o] = f2bf(we2w[K * 64 + N]);
    wx1f[o] = f2bf(wx1w[K * 64 + N]);
  }
  for (int o = tid; o < 8192; o += 256) {       // wh1 transposed [n][k]
    const int k = o & 127, n = o >> 7;
    wh1Ts[o] = f2bf(wh1w[k * 64 + n]);
  }
  for (int o = tid; o < 4096; o += 256) {       // wh2 transposed [n][k]
    const int k = o & 63, n = o >> 6;
    wh2Ts[o] = f2bf(wh2w[k * 64 + n]);
  }
  for (int i = tid; i < 64; i += 256) {
    we1bs[i] = we1b[i]; we2bs[i] = we2b[i]; wx1bs[i] = wx1b[i];
    wh1bs[i] = wh1b[i]; wh2bs[i] = wh2b[i]; we1ls[i] = we1w[128 * 64 + i];
  }
  for (int i = tid; i < 128; i += 256) { wx2s[i] = wx2w[i]; was[i] = waw[i]; }
  if (tid < 2) { wx2bs[tid] = wx2b[tid]; wabs[tid] = wab[tid]; }

  // ---- per-wave node staging ----
  const int wave = tid >> 5, lane = tid & 31;
  char* wbase = smem + OFF_WAVE + wave * PERWAVE;
  unsigned short* buf  = (unsigned short*)(wbase);          // 32x64 bf16 tile, A-frag order
  unsigned short* hibf = (unsigned short*)(wbase + 4096);   // h_i broadcast frags (64 us)
  float* his  = (float*)(wbase + 4224);                     // h_i f32 (64)
  float* dss  = (float*)(wbase + 4480);                     // dist (32)
  float* mis  = (float*)(wbase + 4608);                     // m_i (64)
  float* t1s  = (float*)(wbase + 4864);                     // hidden temp (64)
  float* reds = (float*)(wbase + 5120);                     // reductions (96)

  const int node = blockIdx.x * WAVES + wave;
  for (int i = lane; i < HH; i += 32) { his[i] = hst[(size_t)node * HH + i]; mis[i] = 0.f; }
#pragma unroll
  for (int t = 0; t < 2; ++t) {                             // h_i in A-frag element order
    const int idx = lane + 32 * t;
    const int i = idx & 15, hp = (idx >> 4) & 1, fc = idx >> 5;
    const int feat = fc * 32 + ((i >> 3) << 4) + (i & 7) + hp * 8;
    hibf[idx] = f2bf(hst[(size_t)node * HH + feat]);
  }
  {
    const float* rp = rela + (size_t)(node * KNB + lane) * 6;
    const float dx = rp[0], dy = rp[1];
    dss[lane] = sqrtf(dx * dx + dy * dy);
  }
  {   // h_neigh: coalesced float4 loads -> v4us stores into A-frag layout
    const float4* hp4 = (const float4*)(hneigh + (size_t)node * KNB * HH);
#pragma unroll
    for (int v = 0; v < 16; ++v) {
      const int idx4 = v * 32 + lane;           // 0..511
      const float4 d = hp4[idx4];
      const int e = idx4 >> 4, f0 = (idx4 & 15) << 2;
      v4us st;
      st[0] = f2bf(d.x); st[1] = f2bf(d.y); st[2] = f2bf(d.z); st[3] = f2bf(d.w);
      *(v4us*)(buf + fragAddr(e, f0)) = st;
    }
  }
  if (node + 1 < NODES)
    __builtin_prefetch(hneigh + (size_t)(node + 1) * KNB * HH, 0, 1);
  __syncthreads();

  const int N0 = lane & 15, half = lane >> 4;
  v8f acc[2][4];

  // ================= GEMM1: [32 x 129] @ we1 -> m1 =================
#pragma unroll
  for (int mt = 0; mt < 2; ++mt)
#pragma unroll
    for (int nt = 0; nt < 4; ++nt)
#pragma unroll
      for (int r = 0; r < 8; ++r) {
        const int Nc = nt * 16 + N0;
        const int Mc = mt * 16 + r + 8 * half;
        acc[mt][nt][r] = we1bs[Nc] + dss[Mc] * we1ls[Nc];   // bias + dist column
      }
  {
    const v16bf aHi0 = *(const v16bf*)(hibf + half * 16);
    const v16bf aHi1 = *(const v16bf*)(hibf + (2 + half) * 16);
#pragma unroll
    for (int c = 0; c < 4; ++c) {
      v16bf a0, a1;
      if (c == 0)      { a0 = aHi0; a1 = aHi0; }
      else if (c == 1) { a0 = aHi1; a1 = aHi1; }
      else {
        a0 = *(const v16bf*)(buf + ((0 * 2 + (c - 2)) * 32 + lane) * 16);
        a1 = *(const v16bf*)(buf + ((1 * 2 + (c - 2)) * 32 + lane) * 16);
      }
#pragma unroll
      for (int nt = 0; nt < 4; ++nt) {
        const v16bf bm = *(const v16bf*)(we1f + ((c * 4 + nt) * 32 + lane) * 16);
        acc[0][nt] = __builtin_amdgcn_wmma_f32_16x16x32_bf16(false, a0, false, bm, (short)0, acc[0][nt], false, false);
        acc[1][nt] = __builtin_amdgcn_wmma_f32_16x16x32_bf16(false, a1, false, bm, (short)0, acc[1][nt], false, false);
      }
    }
  }
  // silu -> bf16 -> buf (A-frag layout; scattered b16 stores, no RAW waits)
#pragma unroll
  for (int mt = 0; mt < 2; ++mt)
#pragma unroll
    for (int nt = 0; nt < 4; ++nt)
#pragma unroll
      for (int r = 0; r < 8; ++r) {
        const int Mc = mt * 16 + r + 8 * half, Nc = nt * 16 + N0;
        buf[fragAddr(Mc, Nc)] = f2bf(silu(acc[mt][nt][r]));
      }

  // ================= GEMM2: m1 @ we2 -> m_ij =================
#pragma unroll
  for (int mt = 0; mt < 2; ++mt)
#pragma unroll
    for (int nt = 0; nt < 4; ++nt)
#pragma unroll
      for (int r = 0; r < 8; ++r) acc[mt][nt][r] = we2bs[nt * 16 + N0];
#pragma unroll
  for (int c = 0; c < 2; ++c) {
    const v16bf a0 = *(const v16bf*)(buf + ((0 * 2 + c) * 32 + lane) * 16);
    const v16bf a1 = *(const v16bf*)(buf + ((1 * 2 + c) * 32 + lane) * 16);
#pragma unroll
    for (int nt = 0; nt < 4; ++nt) {
      const v16bf bm = *(const v16bf*)(we2f + ((c * 4 + nt) * 32 + lane) * 16);
      acc[0][nt] = __builtin_amdgcn_wmma_f32_16x16x32_bf16(false, a0, false, bm, (short)0, acc[0][nt], false, false);
      acc[1][nt] = __builtin_amdgcn_wmma_f32_16x16x32_bf16(false, a1, false, bm, (short)0, acc[1][nt], false, false);
    }
  }
  {
    float ps[4] = {0.f, 0.f, 0.f, 0.f};
#pragma unroll
    for (int mt = 0; mt < 2; ++mt)
#pragma unroll
      for (int nt = 0; nt < 4; ++nt)
#pragma unroll
        for (int r = 0; r < 8; ++r) {
          const float v = silu(acc[mt][nt][r]);
          ps[nt] += v;
          const int Mc = mt * 16 + r + 8 * half, Nc = nt * 16 + N0;
          buf[fragAddr(Mc, Nc)] = f2bf(v);
        }
#pragma unroll
    for (int nt = 0; nt < 4; ++nt) atomicAdd(&mis[nt * 16 + N0], ps[nt]); // ds_add_f32
  }

  // ================= GEMM3: m_ij @ wx1 -> x1 =================
#pragma unroll
  for (int mt = 0; mt < 2; ++mt)
#pragma unroll
    for (int nt = 0; nt < 4; ++nt)
#pragma unroll
      for (int r = 0; r < 8; ++r) acc[mt][nt][r] = wx1bs[nt * 16 + N0];
#pragma unroll
  for (int c = 0; c < 2; ++c) {
    const v16bf a0 = *(const v16bf*)(buf + ((0 * 2 + c) * 32 + lane) * 16);
    const v16bf a1 = *(const v16bf*)(buf + ((1 * 2 + c) * 32 + lane) * 16);
#pragma unroll
    for (int nt = 0; nt < 4; ++nt) {
      const v16bf bm = *(const v16bf*)(wx1f + ((c * 4 + nt) * 32 + lane) * 16);
      acc[0][nt] = __builtin_amdgcn_wmma_f32_16x16x32_bf16(false, a0, false, bm, (short)0, acc[0][nt], false, false);
      acc[1][nt] = __builtin_amdgcn_wmma_f32_16x16x32_bf16(false, a1, false, bm, (short)0, acc[1][nt], false, false);
    }
  }
#pragma unroll
  for (int mt = 0; mt < 2; ++mt)
#pragma unroll
    for (int nt = 0; nt < 4; ++nt)
#pragma unroll
      for (int r = 0; r < 8; ++r) {
        const int Mc = mt * 16 + r + 8 * half, Nc = nt * 16 + N0;
        buf[fragAddr(Mc, Nc)] = f2bf(silu(acc[mt][nt][r]));
      }

  // ================= fx = silu(x1 @ wx2 + b); masked-mean agg =================
  {
    const int e = lane;
    float f0 = wx2bs[0], f1 = wx2bs[1];
#pragma unroll
    for (int fc = 0; fc < 2; ++fc)
#pragma unroll
      for (int hp = 0; hp < 2; ++hp) {
        BF16x16 u;
        u.v = *(const v16bf*)(buf + (((e >> 4) * 2 + fc) * 32 + hp * 16 + (e & 15)) * 16);
#pragma unroll
        for (int i = 0; i < 16; ++i) {
          const int K = fc * 32 + ((i >> 3) << 4) + (i & 7) + hp * 8;
          const float xv = bf2f(u.s[i]);
          f0 += xv * wx2s[2 * K + 0];
          f1 += xv * wx2s[2 * K + 1];
        }
      }
    f0 = silu(f0); f1 = silu(f1);
    const float* rp = rela + (size_t)(node * KNB + e) * 6;
    const bool mk = mask[(size_t)node * KNB + e] != 0;
    reds[e]      = mk ? rp[0] * f0 : 0.f;
    reds[32 + e] = mk ? rp[1] * f1 : 0.f;
    reds[64 + e] = mk ? 1.f : 0.f;
  }
  if (lane == 0) {
    float sx = 0.f, sy = 0.f, sc = 0.f;
    for (int e = 0; e < KNB; ++e) { sx += reds[e]; sy += reds[32 + e]; sc += reds[64 + e]; }
    const float inv = 1.f / (sc + 1e-6f);
    reds[0] = sx * inv; reds[1] = sy * inv;
  }

  // ================= node kinematics: a/v/x =================
  if (lane < 2) {
    float s = wabs[lane];
    for (int k3 = 0; k3 < HH; ++k3) s += his[k3] * was[2 * k3 + lane];
    const size_t pb = (size_t)node * 6;
    const float an = silu(s) * ped[pb + 4 + lane] + reds[lane];
    const float vn = ped[pb + 2 + lane] + an;
    const float xn = ped[pb + 0 + lane] + vn;
    out6[pb + 0 + lane] = xn;
    out6[pb + 2 + lane] = vn;
    out6[pb + 4 + lane] = an;
  }

  // ================= h_new = h + f_h(cat(h, m_i)) =================
#pragma unroll
  for (int t = 0; t < 2; ++t) {
    const int n1 = lane + 32 * t;
    float s = wh1bs[n1];
    const unsigned short* wrow = wh1Ts + n1 * 128;
#pragma unroll
    for (int c8 = 0; c8 < 8; ++c8) {
      const v8us w8 = *(const v8us*)(wrow + c8 * 8);
#pragma unroll
      for (int j = 0; j < 8; ++j) s += his[c8 * 8 + j] * bf2f(w8[j]);
    }
#pragma unroll
    for (int c8 = 0; c8 < 8; ++c8) {
      const v8us w8 = *(const v8us*)(wrow + 64 + c8 * 8);
#pragma unroll
      for (int j = 0; j < 8; ++j) s += mis[c8 * 8 + j] * bf2f(w8[j]);
    }
    t1s[n1] = silu(s);
  }
#pragma unroll
  for (int t = 0; t < 2; ++t) {
    const int n1 = lane + 32 * t;
    float s = wh2bs[n1];
    const unsigned short* wrow = wh2Ts + n1 * 64;
#pragma unroll
    for (int c8 = 0; c8 < 8; ++c8) {
      const v8us w8 = *(const v8us*)(wrow + c8 * 8);
#pragma unroll
      for (int j = 0; j < 8; ++j) s += t1s[c8 * 8 + j] * bf2f(w8[j]);
    }
    hout[(size_t)node * HH + n1] = hst[(size_t)node * HH + n1] + s;
  }
}

extern "C" void kernel_launch(void* const* d_in, const int* in_sizes, int n_in,
                              void* d_out, int out_size, void* d_ws, size_t ws_size,
                              hipStream_t stream) {
  (void)in_sizes; (void)n_in; (void)out_size; (void)d_ws; (void)ws_size;
  const float* ped    = (const float*)d_in[0];
  const float* hst    = (const float*)d_in[1];
  const float* hneigh = (const float*)d_in[2];
  const float* rela   = (const float*)d_in[3];
  const unsigned char* mask = (const unsigned char*)d_in[4];
  const float* we1w = (const float*)d_in[5];  const float* we1b = (const float*)d_in[6];
  const float* we2w = (const float*)d_in[7];  const float* we2b = (const float*)d_in[8];
  const float* wx1w = (const float*)d_in[9];  const float* wx1b = (const float*)d_in[10];
  const float* wx2w = (const float*)d_in[11]; const float* wx2b = (const float*)d_in[12];
  const float* waw  = (const float*)d_in[13]; const float* wab  = (const float*)d_in[14];
  const float* wh1w = (const float*)d_in[15]; const float* wh1b = (const float*)d_in[16];
  const float* wh2w = (const float*)d_in[17]; const float* wh2b = (const float*)d_in[18];

  float* out6 = (float*)d_out;                 // [BS,N,6]
  float* hout = out6 + (size_t)NODES * 6;      // [BS,N,64]

  (void)hipFuncSetAttribute((const void*)egnn_fused,
                            hipFuncAttributeMaxDynamicSharedMemorySize, SMEM_TOTAL);

  dim3 grid(NODES / WAVES);  // 2048 blocks, 8 waves (nodes) each
  dim3 block(256);
  egnn_fused<<<grid, block, SMEM_TOTAL, stream>>>(
      ped, hst, hneigh, rela, mask,
      we1w, we1b, we2w, we2b, wx1w, wx1b, wx2w, wx2b,
      waw, wab, wh1w, wh1b, wh2w, wh2b, out6, hout);
}
